// SutraV03_89988154786024
// MI455X (gfx1250) — compile-verified, matrix operand order checked by reading, not run on hardware
//
#include <hip/hip_runtime.h>
#include <hip/hip_bf16.h>

// ---------------------------------------------------------------------------
// MI455X (gfx1250). Dense math via V_WMMA_F32_16X16X32_BF16 (wave32).
// GEMM: 64x128 block tile, 8 waves x (32x32) = 4 WMMA/wave/K-step,
// double-buffered LDS staged with GLOBAL_LOAD_ASYNC_TO_LDS_B128 (ASYNCcnt),
// weights pre-transposed to NxK so both tiles stage as contiguous b128.
// ---------------------------------------------------------------------------

typedef __bf16 bf16;
typedef __bf16 v16bf __attribute__((ext_vector_type(16)));
typedef float  v8f   __attribute__((ext_vector_type(8)));

union FragBF { v16bf v; uint4 u[2]; };

// ---------------- problem constants ----------------
#define Bq    16
#define Tq    4096
#define NTOK  65536      // B*T
#define NP    512        // patches per batch row
#define NROWS 8192       // B*NP
#define DIM_  512
#define D2_   1024
#define VOC   256

// ---------------- workspace layout (bytes) ----------------
static constexpr size_t OFF_LOCAL = 0;                           // f32 65536x512
static constexpr size_t OFF_H     = OFF_LOCAL + 134217728ull;    // f32 65536x512
static constexpr size_t OFF_HN    = OFF_H     + 134217728ull;    // bf16 65536x512
static constexpr size_t OFF_T1    = OFF_HN    + 67108864ull;     // bf16 65536x1024
static constexpr size_t OFF_SMEAN = OFF_T1    + 134217728ull;    // bf16 8192x512
static constexpr size_t OFF_HST   = OFF_SMEAN + 8388608ull;      // f32 8192x512
static constexpr size_t OFF_MIN   = OFF_HST   + 16777216ull;     // bf16 32768x1024
static constexpr size_t OFF_MSGT  = OFF_MIN   + 67108864ull;     // bf16 32768x512
static constexpr size_t OFF_MSGS  = OFF_MSGT  + 33554432ull;     // f32 32768x512
static constexpr size_t OFF_UIN   = OFF_MSGS  + 67108864ull;     // bf16 8192x1024
static constexpr size_t OFF_UPDT  = OFF_UIN   + 16777216ull;     // bf16 8192x512
static constexpr size_t OFF_NEWH  = OFF_UPDT  + 8388608ull;      // f32 8192x512
static constexpr size_t OFF_H2    = OFF_NEWH  + 16777216ull;     // f32 8192x512
static constexpr size_t OFF_OUT   = OFF_H2    + 16777216ull;     // f32 8192x512
static constexpr size_t OFF_MEANH = OFF_OUT   + 16777216ull;     // f32 16x512
static constexpr size_t OFF_STATE = OFF_MEANH + 32768ull;        // f32 scalars
static constexpr size_t OFF_MO    = OFF_STATE + 1024ull;         // bf16 8192x512
static constexpr size_t OFF_Q     = OFF_MO    + 8388608ull;      // f32 8192x512
static constexpr size_t OFF_K     = OFF_Q     + 16777216ull;     // f32
static constexpr size_t OFF_V     = OFF_K     + 16777216ull;     // f32
static constexpr size_t OFF_RETD  = OFF_V     + 16777216ull;     // bf16 8192x512
static constexpr size_t OFF_RETR  = OFF_RETD  + 8388608ull;      // f32 8192x512
static constexpr size_t OFF_BCIN  = OFF_RETR  + 16777216ull;     // bf16 8192x512
static constexpr size_t OFF_BROAD = OFF_BCIN  + 8388608ull;      // f32 8192x512
static constexpr size_t OFF_FIN   = OFF_BROAD + 16777216ull;     // bf16 65536x512
// weights, stored TRANSPOSED (N x K) in bf16:
static constexpr size_t OFF_WPP1  = OFF_FIN   + 67108864ull;     // 1024x512
static constexpr size_t OFF_WPP2  = OFF_WPP1  + 1048576ull;      // 512x1024
static constexpr size_t OFF_WSUM  = OFF_WPP2  + 1048576ull;      // 512x512
static constexpr size_t OFF_WMSG1 = OFF_WSUM  + 524288ull;       // 512x1024
static constexpr size_t OFF_WMSG2 = OFF_WMSG1 + 1048576ull;      // 512x512
static constexpr size_t OFF_WUPD1 = OFF_WMSG2 + 524288ull;       // 512x1024
static constexpr size_t OFF_WUPD2 = OFF_WUPD1 + 1048576ull;      // 512x512
static constexpr size_t OFF_WQ    = OFF_WUPD2 + 524288ull;
static constexpr size_t OFF_WK    = OFF_WQ    + 524288ull;
static constexpr size_t OFF_WV    = OFF_WK    + 524288ull;
static constexpr size_t OFF_WO    = OFF_WV    + 524288ull;
static constexpr size_t OFF_WBC   = OFF_WO    + 524288ull;
static constexpr size_t OFF_WHEAD = OFF_WBC   + 524288ull;       // 256x512

// ---------------- helpers ----------------
__device__ __forceinline__ float blockReduce256(float v, float* buf) {
    const int tid = threadIdx.x;
    buf[tid] = v; __syncthreads();
#pragma unroll
    for (int s = 128; s > 0; s >>= 1) {
        if (tid < s) buf[tid] += buf[tid + s];
        __syncthreads();
    }
    float r = buf[0]; __syncthreads();
    return r;
}

__device__ __forceinline__ float gelu_exact(float x) {
    return 0.5f * x * (1.0f + erff(x * 0.70710678118654752f));
}

// async global->LDS 128-bit copy (tracked by ASYNCcnt)
__device__ __forceinline__ void async_b128(unsigned lds_off, const void* gaddr) {
    asm volatile("global_load_async_to_lds_b128 %0, %1, off"
                 :: "v"(lds_off), "v"(gaddr) : "memory");
}
__device__ __forceinline__ void wait_async_all() {
    asm volatile("s_wait_asynccnt 0x0" ::: "memory");
}
__device__ __forceinline__ void wait_async_keep3() {
    asm volatile("s_wait_asynccnt 0x3" ::: "memory");
}

// ---------------- WMMA bf16 GEMM: out = epi(A @ Wt^T + bias [+res]) -------
// A: MxK bf16 row-major.  Wt: NxK bf16 row-major (pre-transposed weight).
// Block tile 64(M) x 128(N), BK=32, 8 waves each computing 32x32.
__global__ void __launch_bounds__(256)
gemm_bf16_kernel(const bf16* __restrict__ A, const bf16* __restrict__ Wt,
                 const float* __restrict__ bias, const float* __restrict__ res,
                 float* __restrict__ outf, bf16* __restrict__ outb,
                 int M, int N, int K, int doGelu)
{
    __shared__ __align__(16) bf16 As[2][64 * 32];    // [m][k]
    __shared__ __align__(16) bf16 Bs[2][128 * 32];   // [n][k]
    const int tid  = threadIdx.x;
    const int m0   = blockIdx.y * 64;
    const int n0   = blockIdx.x * 128;
    const int lane = tid & 31;
    const int wave = tid >> 5;
    const int wm   = wave & 1;          // 2 waves along M (32 rows each)
    const int wn   = wave >> 1;         // 4 waves along N (32 cols each)
    const int half = lane >> 4;
    const int r    = lane & 15;

    v8f acc[2][2] = {};

    // staging coordinates: one b128 for A, two for B per thread per stage
    const int srow = tid >> 2;          // 0..63
    const int scol = (tid & 3) * 8;     // 0..24 step 8

    const bf16* Agp = A  + (size_t)(m0 + srow) * K + scol;
    const bf16* Bgp = Wt + (size_t)(n0 + srow) * K + scol;
    const bf16* Cgp = Wt + (size_t)(n0 + 64 + srow) * K + scol;

    const unsigned aoff = (unsigned)(size_t)(&As[0][srow * 32 + scol]);
    const unsigned boff = (unsigned)(size_t)(&Bs[0][srow * 32 + scol]);
    const unsigned abuf = (unsigned)(64 * 32 * 2);    // bytes per As buffer
    const unsigned bbuf = (unsigned)(128 * 32 * 2);   // bytes per Bs buffer

    const int S = K >> 5;
    // prologue: stage 0 into buffer 0
    async_b128(aoff, Agp);
    async_b128(boff, Bgp);
    async_b128(boff + 64 * 32 * 2, Cgp);

    for (int s = 0; s < S; ++s) {
        const int buf = s & 1;
        if (s + 1 < S) {                 // prefetch next stage into other buffer
            const int k1 = (s + 1) << 5;
            async_b128(aoff + (buf ^ 1) * abuf, Agp + k1);
            async_b128(boff + (buf ^ 1) * bbuf, Bgp + k1);
            async_b128(boff + (buf ^ 1) * bbuf + 64 * 32 * 2, Cgp + k1);
            wait_async_keep3();          // current stage landed, next in flight
        } else {
            wait_async_all();
        }
        __syncthreads();

        // fragments: b128 LDS loads, lane half h holds k 8h..8h+7 then 16+8h..23+8h
        const int kb0 = half ? 8 : 0;
        const int kb1 = half ? 24 : 16;
        const bf16* Ab = As[buf];
        const bf16* Bb = Bs[buf];
        FragBF a0, a1, b0, b1;
        {
            const bf16* p = Ab + (wm * 32 + r) * 32;
            a0.u[0] = *(const uint4*)(p + kb0); a0.u[1] = *(const uint4*)(p + kb1);
            p = Ab + (wm * 32 + 16 + r) * 32;
            a1.u[0] = *(const uint4*)(p + kb0); a1.u[1] = *(const uint4*)(p + kb1);
            p = Bb + (wn * 32 + r) * 32;
            b0.u[0] = *(const uint4*)(p + kb0); b0.u[1] = *(const uint4*)(p + kb1);
            p = Bb + (wn * 32 + 16 + r) * 32;
            b1.u[0] = *(const uint4*)(p + kb0); b1.u[1] = *(const uint4*)(p + kb1);
        }
        acc[0][0] = __builtin_amdgcn_wmma_f32_16x16x32_bf16(false, a0.v, false, b0.v,
                                                            (short)0, acc[0][0], false, false);
        acc[0][1] = __builtin_amdgcn_wmma_f32_16x16x32_bf16(false, a0.v, false, b1.v,
                                                            (short)0, acc[0][1], false, false);
        acc[1][0] = __builtin_amdgcn_wmma_f32_16x16x32_bf16(false, a1.v, false, b0.v,
                                                            (short)0, acc[1][0], false, false);
        acc[1][1] = __builtin_amdgcn_wmma_f32_16x16x32_bf16(false, a1.v, false, b1.v,
                                                            (short)0, acc[1][1], false, false);
        __syncthreads();
    }

    // epilogue: C/D layout — VGPR rr, lane l: M=(l<16? rr : rr+8), N=l&15
    const int nlo = lane & 15;
#pragma unroll
    for (int i = 0; i < 2; ++i) {
        const int mbase = m0 + wm * 32 + i * 16 + (half ? 8 : 0);
#pragma unroll
        for (int j = 0; j < 2; ++j) {
            const int gcol = n0 + wn * 32 + j * 16 + nlo;
#pragma unroll
            for (int rr = 0; rr < 8; ++rr) {
                size_t idx = (size_t)(mbase + rr) * N + gcol;
                float v = acc[i][j][rr];
                if (bias)   v += bias[gcol];
                if (res)    v += res[idx];
                if (doGelu) v  = gelu_exact(v);
                if (outf) outf[idx] = v;
                if (outb) outb[idx] = (bf16)v;
            }
        }
    }
}

// ---------------- embed + pos + LN (h fp32, hn bf16) ----------------
__global__ void __launch_bounds__(256)
embed_ln_kernel(const int* __restrict__ x, const float* __restrict__ emb,
                const float* __restrict__ pos, const float* __restrict__ g,
                const float* __restrict__ bb, float* __restrict__ h,
                bf16* __restrict__ hn)
{
    __shared__ float red[256];
    const int i   = blockIdx.x;      // flat token
    const int tid = threadIdx.x;
    const int tok = x[i];
    const int pr  = i & 7;           // position within patch
    float v0 = emb[(size_t)tok * DIM_ + tid]        + pos[pr * DIM_ + tid];
    float v1 = emb[(size_t)tok * DIM_ + 256 + tid]  + pos[pr * DIM_ + 256 + tid];
    float m   = blockReduce256(v0 + v1, red) * (1.0f / DIM_);
    float d0 = v0 - m, d1 = v1 - m;
    float var = blockReduce256(d0 * d0 + d1 * d1, red) * (1.0f / DIM_);
    float is  = rsqrtf(var + 1e-5f);
    size_t base = (size_t)i * DIM_;
    h[base + tid]        = v0;
    h[base + 256 + tid]  = v1;
    hn[base + tid]       = (bf16)(d0 * is * g[tid] + bb[tid]);
    hn[base + 256 + tid] = (bf16)(d1 * is * g[256 + tid] + bb[256 + tid]);
}

// ---------------- patch mean -> bf16 ----------------
__global__ void __launch_bounds__(256)
patch_mean_kernel(const float* __restrict__ local, bf16* __restrict__ sm)
{
    const int row = blockIdx.x;
    const int tid = threadIdx.x;
#pragma unroll
    for (int q = 0; q < 2; ++q) {
        int d = tid + q * 256;
        float s = 0.f;
#pragma unroll
        for (int p = 0; p < 8; ++p) s += local[((size_t)row * 8 + p) * DIM_ + d];
        sm[(size_t)row * DIM_ + d] = (bf16)(s * 0.125f);
    }
}

// ---------------- init: zero msg-out accumulator + halting state ----------
__global__ void __launch_bounds__(256)
init_state_kernel(float* __restrict__ out, float* __restrict__ state)
{
    size_t i = (size_t)blockIdx.x * 256 + threadIdx.x;
    out[i] = 0.f;
    if (blockIdx.x == 0 && threadIdx.x < 64)
        state[threadIdx.x] = (threadIdx.x < 16) ? 1.0f : 0.0f;  // rem=1, rest 0
}

// ---------------- gather window neighbors -> m_in bf16 --------------------
__global__ void __launch_bounds__(256)
gather_min_kernel(const float* __restrict__ hs, bf16* __restrict__ mi)
{
    const int row = blockIdx.x;   // b*NP+n
    const int j   = blockIdx.y;   // 0..3 (offset j-3)
    const int n   = row & (NP - 1);
    const int b   = row >> 9;
    int idx = n + j - 3; if (idx < 0) idx = 0;
    size_t so  = (size_t)row * DIM_;
    size_t no  = ((size_t)(b * NP + idx)) * DIM_;
    size_t dst = ((size_t)row * 4 + j) * D2_;
    const int tid = threadIdx.x;
#pragma unroll
    for (int q = 0; q < 2; ++q) {
        int d = tid + q * 256;
        mi[dst + d]        = (bf16)hs[so + d];
        mi[dst + 512 + d]  = (bf16)hs[no + d];
    }
}

// ---------------- masked mean over window + concat -> u_in bf16 -----------
__global__ void __launch_bounds__(256)
agg_uin_kernel(const float* __restrict__ msgs, const float* __restrict__ hs,
               bf16* __restrict__ uin)
{
    const int row = blockIdx.x;
    const int n   = row & (NP - 1);
    const int cnt = (n < 3 ? n : 3) + 1;
    const float inv = 1.0f / (float)cnt;
    const int tid = threadIdx.x;
#pragma unroll
    for (int q = 0; q < 2; ++q) {
        int d = tid + q * 256;
        float s = 0.f;
#pragma unroll
        for (int j = 0; j < 4; ++j)
            if (n + j - 3 >= 0) s += msgs[(((size_t)row * 4) + j) * DIM_ + d];
        uin[(size_t)row * D2_ + d]       = (bf16)hs[(size_t)row * DIM_ + d];
        uin[(size_t)row * D2_ + 512 + d] = (bf16)(s * inv);
    }
}

// ---------------- row LN (fp32 in/out) ----------------
__global__ void __launch_bounds__(256)
rowln_kernel(const float* __restrict__ X, const float* __restrict__ g,
             const float* __restrict__ bb, float* __restrict__ Y)
{
    __shared__ float red[256];
    const int row = blockIdx.x;
    const int tid = threadIdx.x;
    size_t base = (size_t)row * DIM_;
    float v0 = X[base + tid], v1 = X[base + 256 + tid];
    float m   = blockReduce256(v0 + v1, red) * (1.0f / DIM_);
    float d0 = v0 - m, d1 = v1 - m;
    float var = blockReduce256(d0 * d0 + d1 * d1, red) * (1.0f / DIM_);
    float is  = rsqrtf(var + 1e-5f);
    Y[base + tid]       = d0 * is * g[tid] + bb[tid];
    Y[base + 256 + tid] = d1 * is * g[256 + tid] + bb[256 + tid];
}

// ---------------- mean over patches per batch: (B, D) ----------------
__global__ void __launch_bounds__(128)
meanh_kernel(const float* __restrict__ h2, float* __restrict__ mh)
{
    const int b = blockIdx.x;
    const int d = blockIdx.y * 128 + threadIdx.x;
    float s = 0.f;
    for (int n = 0; n < NP; ++n) s += h2[((size_t)(b * NP + n)) * DIM_ + d];
    mh[b * DIM_ + d] = s * (1.0f / NP);
}

// ---------------- halting scalars (ACT bookkeeping) ----------------
__global__ void __launch_bounds__(512)
halt_kernel(const float* __restrict__ mh, const float* __restrict__ hw,
            const float* __restrict__ hb, float* __restrict__ st, int t)
{
    __shared__ float red[512];
    __shared__ float hpArr[16];
    const int tid = threadIdx.x;
    for (int b = 0; b < Bq; ++b) {
        red[tid] = mh[b * DIM_ + tid] * hw[tid];
        __syncthreads();
        for (int s = 256; s > 0; s >>= 1) {
            if (tid < s) red[tid] += red[tid + s];
            __syncthreads();
        }
        if (tid == 0) hpArr[b] = red[0];
        __syncthreads();
    }
    if (tid == 0) {
        float halted = st[33];
        float af = (halted != 0.f) ? 0.f : 1.f;
        float hp[16]; float hpm = 0.f;
        for (int b = 0; b < Bq; ++b) {
            float z = hpArr[b] + hb[0];
            float p = 1.f / (1.f + expf(-z));
            if (t == 7) p = 1.f;
            hp[b] = p; hpm += p;
        }
        hpm *= (1.f / Bq);
        float maxrem = -1.f;
        for (int b = 0; b < Bq; ++b) {
            float rem = st[b];
            st[16 + b] = af * rem * hp[b];                        // out coeff
            float remn = (halted != 0.f) ? rem : rem * (1.f - hp[b]);
            st[b] = remn;
            maxrem = fmaxf(maxrem, remn);
        }
        float pg = 0.2f * powf(0.8f, (float)t);
        st[32] += af * (pg * (logf(pg) - logf(hpm + 1e-8f)));     // kl
        st[34] = af;                                              // active flag
        if (maxrem < 0.01f) st[33] = 1.f;                         // halted
    }
}

// ---------------- out += coef[b]*h2; hstate = active? h2 : hstate ---------
__global__ void __launch_bounds__(256)
apply_kernel(const float* __restrict__ h2, float* __restrict__ out,
             float* __restrict__ hs, const float* __restrict__ st)
{
    const int row = blockIdx.x;
    const int b   = row >> 9;
    const float c   = st[16 + b];
    const float act = st[34];
    const int tid = threadIdx.x;
#pragma unroll
    for (int q = 0; q < 2; ++q) {
        size_t i = (size_t)row * DIM_ + tid + q * 256;
        float hv = h2[i];
        out[i] += c * hv;
        if (act != 0.f) hs[i] = hv;
    }
}

// ---------------- fp32 -> bf16 convert ----------------
__global__ void __launch_bounds__(256)
f2b_kernel(const float* __restrict__ src, bf16* __restrict__ dst, int n)
{
    int i = blockIdx.x * 256 + threadIdx.x;
    if (i < n) dst[i] = (bf16)src[i];
}

// ---------------- fp32 KxN -> bf16 NxK (weight transpose) ----------------
__global__ void __launch_bounds__(256)
w2bt_kernel(const float* __restrict__ src, bf16* __restrict__ dst, int K, int N)
{
    int i = blockIdx.x * 256 + threadIdx.x;
    if (i < K * N) {
        int k = i / N, n = i - k * N;
        dst[(size_t)n * K + k] = (bf16)src[i];
    }
}

// ---------------- a+b -> bf16 ----------------
__global__ void __launch_bounds__(256)
add_f2b_kernel(const float* __restrict__ a, const float* __restrict__ b,
               bf16* __restrict__ dst, int n)
{
    int i = blockIdx.x * 256 + threadIdx.x;
    if (i < n) dst[i] = (bf16)(a[i] + b[i]);
}

// ---------------- causal top-4 retrieval attention ----------------
__global__ void __launch_bounds__(256)
retrieval_kernel(const float* __restrict__ q, const float* __restrict__ k,
                 const float* __restrict__ v, bf16* __restrict__ rd)
{
    __shared__ float qs[512];
    __shared__ float sc[512];
    __shared__ float wts[4];
    __shared__ int   widx[4];
    const int row = blockIdx.x;            // b*NP+n
    const int n   = row & (NP - 1);
    const int b   = row >> 9;
    const int tid = threadIdx.x;
    qs[tid]       = q[(size_t)row * DIM_ + tid];
    qs[tid + 256] = q[(size_t)row * DIM_ + 256 + tid];
    __syncthreads();
#pragma unroll
    for (int qi = 0; qi < 2; ++qi) {
        int m = tid + qi * 256;
        float s;
        if (m <= n) {
            const float* kp = k + ((size_t)(b * NP + m)) * DIM_;
            float acc = 0.f;
            for (int d = 0; d < DIM_; ++d) acc += qs[d] * kp[d];
            s = acc * 0.04419417382415922f;   // 1/sqrt(512)
        } else s = -1e30f;
        sc[m] = s;
    }
    __syncthreads();
    if (tid == 0) {
        float vals[4];
        for (int kk = 0; kk < 4; ++kk) {      // top-4, earliest-index ties
            float best = -3.0e38f; int bi = 0;
            for (int m = 0; m < NP; ++m) {
                float x = sc[m];
                if (x > best) { best = x; bi = m; }
            }
            vals[kk] = best; widx[kk] = bi; sc[bi] = -3.0e38f;
        }
        float mx = vals[0], se = 0.f, e[4];
        for (int kk = 0; kk < 4; ++kk) { e[kk] = expf(vals[kk] - mx); se += e[kk]; }
        for (int kk = 0; kk < 4; ++kk) wts[kk] = e[kk] / se;
    }
    __syncthreads();
#pragma unroll
    for (int qi = 0; qi < 2; ++qi) {
        int d = tid + qi * 256;
        float acc = 0.f;
#pragma unroll
        for (int kk = 0; kk < 4; ++kk)
            acc += wts[kk] * v[((size_t)(b * NP + widx[kk])) * DIM_ + d];
        rd[(size_t)row * DIM_ + d] = (bf16)acc;
    }
}

// ---------------- final = local + broad (broadcast), LN -> bf16 -----------
__global__ void __launch_bounds__(256)
final_ln_kernel(const float* __restrict__ local, const float* __restrict__ broad,
                const float* __restrict__ g, const float* __restrict__ bb,
                bf16* __restrict__ fin)
{
    __shared__ float red[256];
    const int i   = blockIdx.x;     // flat token
    const int tid = threadIdx.x;
    const int b   = i >> 12;
    const int t   = i & 4095;
    const int np  = t >> 3;
    size_t bo   = ((size_t)(b * NP + np)) * DIM_;
    size_t base = (size_t)i * DIM_;
    float v0 = local[base + tid]       + broad[bo + tid];
    float v1 = local[base + 256 + tid] + broad[bo + 256 + tid];
    float m   = blockReduce256(v0 + v1, red) * (1.0f / DIM_);
    float d0 = v0 - m, d1 = v1 - m;
    float var = blockReduce256(d0 * d0 + d1 * d1, red) * (1.0f / DIM_);
    float is  = rsqrtf(var + 1e-5f);
    fin[base + tid]       = (bf16)(d0 * is * g[tid] + bb[tid]);
    fin[base + 256 + tid] = (bf16)(d1 * is * g[256 + tid] + bb[256 + tid]);
}

// ---------------- KL tail write ----------------
__global__ void kl_write_kernel(const float* __restrict__ st, float* __restrict__ out)
{
    if (threadIdx.x == 0 && blockIdx.x == 0)
        out[(size_t)NTOK * VOC] = st[32] * 0.125f;   // kl / MAX_ROUNDS
}

// ---------------------------------------------------------------------------
extern "C" void kernel_launch(void* const* d_in, const int* in_sizes, int n_in,
                              void* d_out, int out_size, void* d_ws, size_t ws_size,
                              hipStream_t stream)
{
    (void)in_sizes; (void)n_in; (void)out_size; (void)ws_size;
    char* wsb = (char*)d_ws;
    auto F  = [&](size_t off) { return (float*)(wsb + off); };
    auto Bw = [&](size_t off) { return (bf16*)(wsb + off); };

    const int*   x      = (const int*)d_in[0];
    const float* emb    = (const float*)d_in[1];
    const float* pos    = (const float*)d_in[2];
    const float* ppg    = (const float*)d_in[3];
    const float* ppb    = (const float*)d_in[4];
    const float* pp_b1  = (const float*)d_in[6];
    const float* pp_b2  = (const float*)d_in[8];
    const float* sum_b  = (const float*)d_in[10];
    const float* msg_b1 = (const float*)d_in[12];
    const float* msg_b2 = (const float*)d_in[14];
    const float* upd_b1 = (const float*)d_in[16];
    const float* upd_b2 = (const float*)d_in[18];
    const float* halt_w = (const float*)d_in[19];
    const float* halt_b = (const float*)d_in[20];
    const float* mpg    = (const float*)d_in[21];
    const float* mpb    = (const float*)d_in[22];
    const float* q_b    = (const float*)d_in[24];
    const float* k_b    = (const float*)d_in[26];
    const float* v_b    = (const float*)d_in[28];
    const float* o_b    = (const float*)d_in[30];
    const float* bc_b   = (const float*)d_in[32];
    const float* ln_g   = (const float*)d_in[33];
    const float* ln_b   = (const float*)d_in[34];

    auto cvtT = [&](int idx, size_t off, int K, int N) {
        int n = K * N;
        w2bt_kernel<<<(n + 255) / 256, 256, 0, stream>>>((const float*)d_in[idx],
                                                         Bw(off), K, N);
    };
    auto gemm = [&](const bf16* A, const bf16* W, const float* bias, const float* res,
                    float* outf, bf16* outb, int M, int N, int K, int gl) {
        dim3 g(N / 128, M / 64);
        gemm_bf16_kernel<<<g, 256, 0, stream>>>(A, W, bias, res, outf, outb, M, N, K, gl);
    };

    // weights -> bf16, transposed to NxK
    cvtT(5,  OFF_WPP1, 512, 1024);  cvtT(7,  OFF_WPP2, 1024, 512);
    cvtT(9,  OFF_WSUM, 512, 512);   cvtT(11, OFF_WMSG1, 1024, 512);
    cvtT(13, OFF_WMSG2, 512, 512);  cvtT(15, OFF_WUPD1, 1024, 512);
    cvtT(17, OFF_WUPD2, 512, 512);  cvtT(23, OFF_WQ, 512, 512);
    cvtT(25, OFF_WK, 512, 512);     cvtT(27, OFF_WV, 512, 512);
    cvtT(29, OFF_WO, 512, 512);     cvtT(31, OFF_WBC, 512, 512);
    cvtT(35, OFF_WHEAD, 512, 256);

    // embed + pos + LN
    embed_ln_kernel<<<NTOK, 256, 0, stream>>>(x, emb, pos, ppg, ppb, F(OFF_H), Bw(OFF_HN));
    // patch MLP: t1 = gelu(hn@pp_w1+b1) ; local = h + t1@pp_w2+b2
    gemm(Bw(OFF_HN), Bw(OFF_WPP1), pp_b1, nullptr, nullptr, Bw(OFF_T1), NTOK, D2_, DIM_, 1);
    gemm(Bw(OFF_T1), Bw(OFF_WPP2), pp_b2, F(OFF_H), F(OFF_LOCAL), nullptr, NTOK, DIM_, D2_, 0);
    // summaries
    patch_mean_kernel<<<NROWS, 256, 0, stream>>>(F(OFF_LOCAL), Bw(OFF_SMEAN));
    gemm(Bw(OFF_SMEAN), Bw(OFF_WSUM), sum_b, nullptr, F(OFF_HST), nullptr, NROWS, DIM_, DIM_, 0);
    init_state_kernel<<<16384, 256, 0, stream>>>(F(OFF_OUT), F(OFF_STATE));

    // message passing rounds
    for (int t = 0; t < 8; ++t) {
        gather_min_kernel<<<dim3(NROWS, 4), 256, 0, stream>>>(F(OFF_HST), Bw(OFF_MIN));
        gemm(Bw(OFF_MIN), Bw(OFF_WMSG1), msg_b1, nullptr, nullptr, Bw(OFF_MSGT),
             NROWS * 4, DIM_, D2_, 1);
        gemm(Bw(OFF_MSGT), Bw(OFF_WMSG2), msg_b2, nullptr, F(OFF_MSGS), nullptr,
             NROWS * 4, DIM_, DIM_, 0);
        agg_uin_kernel<<<NROWS, 256, 0, stream>>>(F(OFF_MSGS), F(OFF_HST), Bw(OFF_UIN));
        gemm(Bw(OFF_UIN), Bw(OFF_WUPD1), upd_b1, nullptr, nullptr, Bw(OFF_UPDT),
             NROWS, DIM_, D2_, 1);
        gemm(Bw(OFF_UPDT), Bw(OFF_WUPD2), upd_b2, F(OFF_HST), F(OFF_NEWH), nullptr,
             NROWS, DIM_, DIM_, 0);
        rowln_kernel<<<NROWS, 256, 0, stream>>>(F(OFF_NEWH), mpg, mpb, F(OFF_H2));
        meanh_kernel<<<dim3(Bq, 4), 128, 0, stream>>>(F(OFF_H2), F(OFF_MEANH));
        halt_kernel<<<1, 512, 0, stream>>>(F(OFF_MEANH), halt_w, halt_b, F(OFF_STATE), t);
        apply_kernel<<<NROWS, 256, 0, stream>>>(F(OFF_H2), F(OFF_OUT), F(OFF_HST), F(OFF_STATE));
    }

    // retrieval on msg_out
    f2b_kernel<<<16384, 256, 0, stream>>>(F(OFF_OUT), Bw(OFF_MO), NROWS * DIM_);
    gemm(Bw(OFF_MO), Bw(OFF_WQ), q_b, nullptr, F(OFF_Q), nullptr, NROWS, DIM_, DIM_, 0);
    gemm(Bw(OFF_MO), Bw(OFF_WK), k_b, nullptr, F(OFF_K), nullptr, NROWS, DIM_, DIM_, 0);
    gemm(Bw(OFF_MO), Bw(OFF_WV), v_b, nullptr, F(OFF_V), nullptr, NROWS, DIM_, DIM_, 0);
    retrieval_kernel<<<NROWS, 256, 0, stream>>>(F(OFF_Q), F(OFF_K), F(OFF_V), Bw(OFF_RETD));
    gemm(Bw(OFF_RETD), Bw(OFF_WO), o_b, nullptr, F(OFF_RETR), nullptr, NROWS, DIM_, DIM_, 0);

    // broadcast + final head
    add_f2b_kernel<<<16384, 256, 0, stream>>>(F(OFF_OUT), F(OFF_RETR), Bw(OFF_BCIN), NROWS * DIM_);
    gemm(Bw(OFF_BCIN), Bw(OFF_WBC), bc_b, nullptr, F(OFF_BROAD), nullptr, NROWS, DIM_, DIM_, 0);
    final_ln_kernel<<<NTOK, 256, 0, stream>>>(F(OFF_LOCAL), F(OFF_BROAD), ln_g, ln_b, Bw(OFF_FIN));
    gemm(Bw(OFF_FIN), Bw(OFF_WHEAD), nullptr, nullptr, (float*)d_out, nullptr, NTOK, VOC, DIM_, 0);
    kl_write_kernel<<<1, 32, 0, stream>>>(F(OFF_STATE), (float*)d_out);
}
